// Attention_81776177315877
// MI455X (gfx1250) — matrix operand order repaired and checked by reading, not compiled, not run on hardware
//
#include <hip/hip_runtime.h>
#include <hip/hip_bf16.h>

// ---------------- problem constants ----------------
#define CC      256
#define NHEADS  8
#define DHEAD   32
#define NB      8
#define IH      56
#define IW      56
#define NQ      (IH * IW)   // 3136
#define OHW     28
#define NKV     (OHW * OHW) // 784
#define ATT_SCALE 0.17677669529663687f  // 32^-0.5
#define BN_EPS  1e-5f

typedef _Float16 half_t;
typedef _Float16 v16h __attribute__((ext_vector_type(16)));
typedef _Float16 v8h  __attribute__((ext_vector_type(8)));
typedef float    v8f  __attribute__((ext_vector_type(8)));

// ---------------- DPP 16-lane reductions (pure VALU, no LDS) --------------
// WMMA D-layout rows live in 16-lane halves; these controls never cross them.
template <int CTRL>
__device__ __forceinline__ float dpp_mov(float x) {
    return __builtin_bit_cast(float,
        __builtin_amdgcn_update_dpp(0, __builtin_bit_cast(int, x),
                                    CTRL, 0xF, 0xF, true));
}
#define DPP_XOR1  0xB1   // quad_perm(1,0,3,2)
#define DPP_XOR2  0x4E   // quad_perm(2,3,0,1)
#define DPP_HMIRR 0x141  // row_half_mirror
#define DPP_MIRR  0x140  // row_mirror

__device__ __forceinline__ float red_max16(float x) {
    x = fmaxf(x, dpp_mov<DPP_XOR1>(x));
    x = fmaxf(x, dpp_mov<DPP_XOR2>(x));
    x = fmaxf(x, dpp_mov<DPP_HMIRR>(x));
    x = fmaxf(x, dpp_mov<DPP_MIRR>(x));
    return x;
}
__device__ __forceinline__ float red_sum16(float x) {
    x += dpp_mov<DPP_XOR1>(x);
    x += dpp_mov<DPP_XOR2>(x);
    x += dpp_mov<DPP_HMIRR>(x);
    x += dpp_mov<DPP_MIRR>(x);
    return x;
}

// ---------------- WMMA fragment loaders (CDNA5 16x16x32 f16 layouts) ------
__device__ __forceinline__ v16h load_a_rowmajor(const half_t* __restrict__ base,
                                                int row0, int ld, int lane) {
    int m  = lane & 15;
    int kb = (lane & 16) ? 8 : 0;
    const half_t* p = base + (size_t)(row0 + m) * ld + kb;
    v8h lo = *(const v8h*)(p);
    v8h hi = *(const v8h*)(p + 16);
    return __builtin_shufflevector(lo, hi, 0, 1, 2, 3, 4, 5, 6, 7,
                                   8, 9, 10, 11, 12, 13, 14, 15);
}

__device__ __forceinline__ v16h load_b_nmajor(const half_t* __restrict__ base,
                                              int ld, int col0, int lane) {
    int n  = col0 + (lane & 15);
    int kb = (lane & 16) ? 8 : 0;
    const half_t* p = base + (size_t)n * ld + kb;
    v8h lo = *(const v8h*)(p);
    v8h hi = *(const v8h*)(p + 16);
    return __builtin_shufflevector(lo, hi, 0, 1, 2, 3, 4, 5, 6, 7,
                                   8, 9, 10, 11, 12, 13, 14, 15);
}

// ---------------- kernel 1: weight prep (f32 -> f16, BN fold) -------------
__global__ void prep_kernel(const float* __restrict__ wq, const float* __restrict__ wk,
                            const float* __restrict__ wv, const float* __restrict__ pw,
                            const float* __restrict__ qg, const float* __restrict__ qb,
                            const float* __restrict__ qm, const float* __restrict__ qv,
                            const float* __restrict__ kg, const float* __restrict__ kb,
                            const float* __restrict__ km, const float* __restrict__ kv,
                            const float* __restrict__ vg, const float* __restrict__ vb,
                            const float* __restrict__ vm, const float* __restrict__ vv,
                            half_t* __restrict__ wq16, half_t* __restrict__ wk16,
                            half_t* __restrict__ wv16, half_t* __restrict__ pw16,
                            float* __restrict__ bnp) {
    int idx = blockIdx.x * blockDim.x + threadIdx.x;  // 0 .. 65535
    wq16[idx] = (half_t)wq[idx];
    wk16[idx] = (half_t)wk[idx];
    wv16[idx] = (half_t)wv[idx];
    pw16[idx] = (half_t)pw[idx];  // proj stays [o][c]: already N-major for B frags
    if (idx < CC) {
        float s;
        s = qg[idx] * rsqrtf(qv[idx] + BN_EPS);
        bnp[0 * CC + idx] = s; bnp[1 * CC + idx] = qb[idx] - qm[idx] * s;
        s = kg[idx] * rsqrtf(kv[idx] + BN_EPS);
        bnp[2 * CC + idx] = s; bnp[3 * CC + idx] = kb[idx] - km[idx] * s;
        s = vg[idx] * rsqrtf(vv[idx] + BN_EPS);
        bnp[4 * CC + idx] = s; bnp[5 * CC + idx] = vb[idx] - vm[idx] * s;
    }
}

// ---------------- kernel 2: depthwise 3x3 + BN -> f16 [b][n][c] -----------
__global__ __launch_bounds__(256) void dwconv_bn_kernel(
        const float* __restrict__ x, const float* __restrict__ dww,
        const float* __restrict__ dwb, const float* __restrict__ bnsc,
        const float* __restrict__ bnsh, half_t* __restrict__ t,
        int stride, int Wout, int Nout) {
    int c = threadIdx.x;  // one channel per lane -> coalesced x reads/writes
    int ntiles = Nout >> 4;
    int b  = blockIdx.x / ntiles;
    int nt = blockIdx.x % ntiles;
    float w[9];
#pragma unroll
    for (int i = 0; i < 9; ++i) w[i] = dww[c * 9 + i];
    float bias = dwb[c], sc = bnsc[c], sh = bnsh[c];
    const float* xb = x + (size_t)b * NQ * CC;
    half_t* tb = t + (size_t)b * Nout * CC;
    for (int p = 0; p < 16; ++p) {
        int n  = nt * 16 + p;
        int ho = n / Wout, wo = n % Wout;
        int hi0 = ho * stride - 1, wi0 = wo * stride - 1;
        float acc = 0.f;
#pragma unroll
        for (int dh = 0; dh < 3; ++dh) {
            int hi = hi0 + dh;
            if ((unsigned)hi >= (unsigned)IH) continue;
#pragma unroll
            for (int dw = 0; dw < 3; ++dw) {
                int wi = wi0 + dw;
                if ((unsigned)wi >= (unsigned)IW) continue;
                acc += xb[(size_t)(hi * IW + wi) * CC + c] * w[dh * 3 + dw];
            }
        }
        tb[(size_t)n * CC + c] = (half_t)((acc + bias) * sc + sh);
    }
}

// ---------------- kernel 3: pointwise 1x1 GEMM via WMMA -------------------
template <int NBLK>
__global__ __launch_bounds__(128) void pw_gemm_kernel(
        const half_t* __restrict__ W16, const float* __restrict__ bias,
        const half_t* __restrict__ T, half_t* __restrict__ OUT,
        int N, int transposeHeads, float oscale) {
    int lane = threadIdx.x & 31;
    int wid  = blockIdx.x * (blockDim.x >> 5) + (threadIdx.x >> 5);
    int ng   = (N >> 4) / NBLK;  // groups of NBLK n-tiles
    int b    = wid / (16 * ng);
    int rem  = wid - b * 16 * ng;
    int ot   = rem / ng;
    int gi   = rem - ot * ng;
    int nt0  = gi * NBLK;

    const half_t* tbase = T + ((size_t)b * N + nt0 * 16) * CC;
    v8f acc[NBLK];
#pragma unroll
    for (int nb = 0; nb < NBLK; ++nb) acc[nb] = (v8f){};
#pragma unroll
    for (int kc = 0; kc < 8; ++kc) {
        v16h a = load_a_rowmajor(W16 + kc * 32, ot * 16, CC, lane);
#pragma unroll
        for (int nb = 0; nb < NBLK; ++nb) {
            v16h bf = load_b_nmajor(tbase + (size_t)nb * 16 * CC + kc * 32, CC, 0, lane);
            acc[nb] = __builtin_amdgcn_wmma_f32_16x16x32_f16(false, a, false, bf,
                                                             (short)0, acc[nb], false, false);
        }
    }
    int mh = (lane & 16) ? 8 : 0;
    int nl = lane & 15;
#pragma unroll
    for (int nb = 0; nb < NBLK; ++nb) {
        int ncol = (nt0 + nb) * 16 + nl;
#pragma unroll
        for (int r = 0; r < 8; ++r) {
            int orow  = ot * 16 + r + mh;
            float val = (acc[nb][r] + bias[orow]) * oscale;
            if (transposeHeads) {  // [b][h][n][d]
                OUT[(((size_t)b * NHEADS + (orow >> 5)) * N + ncol) * DHEAD + (orow & 31)] =
                    (half_t)val;
            } else {               // [b][o][n] == [b][h][d][n]
                OUT[((size_t)b * CC + orow) * N + ncol] = (half_t)val;
            }
        }
    }
}

// ---------------- kernel 4: flash attention via WMMA ----------------------
// 32 query rows per wave; 32-key chunks with ONE softmax update per chunk.
__device__ __forceinline__ void softmax_update2(v8f sa, v8f sb,
                                                float (&mrow)[8], float (&lrow)[8],
                                                v8f& oa, v8f& ob, half_t (&pl)[16][32],
                                                int mh, int nl) {
#pragma unroll
    for (int r = 0; r < 8; ++r) {
        float mx   = red_max16(fmaxf(sa[r], sb[r]));
        float mnew = fmaxf(mrow[r], mx);
        float corr = __expf(mrow[r] - mnew);
        float pa   = __expf(sa[r] - mnew);
        float pb   = __expf(sb[r] - mnew);
        float rs   = red_sum16(pa + pb);
        lrow[r] = lrow[r] * corr + rs;
        oa[r] *= corr;
        ob[r] *= corr;
        mrow[r] = mnew;
        pl[r + mh][nl]      = (half_t)pa;
        pl[r + mh][16 + nl] = (half_t)pb;
    }
}

// tail: only the low 16 key columns are valid; high half forced to p=0
__device__ __forceinline__ void softmax_update_tail(v8f sa,
                                                    float (&mrow)[8], float (&lrow)[8],
                                                    v8f& oa, v8f& ob, half_t (&pl)[16][32],
                                                    int mh, int nl) {
#pragma unroll
    for (int r = 0; r < 8; ++r) {
        float mx   = red_max16(sa[r]);
        float mnew = fmaxf(mrow[r], mx);
        float corr = __expf(mrow[r] - mnew);
        float pa   = __expf(sa[r] - mnew);
        float rs   = red_sum16(pa);
        lrow[r] = lrow[r] * corr + rs;
        oa[r] *= corr;
        ob[r] *= corr;
        mrow[r] = mnew;
        pl[r + mh][nl]      = (half_t)pa;
        pl[r + mh][16 + nl] = (half_t)0.f;
    }
}

__device__ __forceinline__ v16h load_p_frag(const half_t (&pl)[16][32], int lane) {
    int m  = lane & 15;
    int kb = (lane & 16) ? 8 : 0;
    const half_t* p = &pl[m][kb];
    v8h lo = *(const v8h*)(p);
    v8h hi = *(const v8h*)(p + 16);
    return __builtin_shufflevector(lo, hi, 0, 1, 2, 3, 4, 5, 6, 7,
                                   8, 9, 10, 11, 12, 13, 14, 15);
}

__global__ __launch_bounds__(64) void attention_kernel(
        const half_t* __restrict__ qh, const half_t* __restrict__ kh,
        const half_t* __restrict__ vh, half_t* __restrict__ obuf) {
    __shared__ half_t plds[2][2][16][32];  // [wave][mblk][row][k]
    int lane  = threadIdx.x & 31;
    int w     = threadIdx.x >> 5;
    int itile = blockIdx.x % 49;
    int bh    = blockIdx.x / 49;
    int h     = bh & 7, b = bh >> 3;
    int i0    = itile * 64 + w * 32;

    const half_t* qbase = qh + ((size_t)bh * NQ + i0) * DHEAD;
    const half_t* kbase = kh + (size_t)bh * NKV * DHEAD;   // [j][d]
    const half_t* vbase = vh + (size_t)bh * DHEAD * NKV;   // [d][j]

    v16h qa0 = load_a_rowmajor(qbase, 0, DHEAD, lane);
    v16h qa1 = load_a_rowmajor(qbase, 16, DHEAD, lane);

    float mrow[2][8], lrow[2][8];
#pragma unroll
    for (int mb = 0; mb < 2; ++mb)
#pragma unroll
        for (int r = 0; r < 8; ++r) { mrow[mb][r] = -1e30f; lrow[mb][r] = 0.f; }
    v8f o00 = {}, o01 = {}, o10 = {}, o11 = {};
    int mh = (lane & 16) ? 8 : 0;
    int nl = lane & 15;
    v8f zc = {};

    for (int it = 0; it < 24; ++it) {  // 32-key chunks, j = it*32 .. it*32+31
        v16h kfa = load_b_nmajor(kbase, DHEAD, it * 32, lane);
        v16h kfb = load_b_nmajor(kbase, DHEAD, it * 32 + 16, lane);
        v8f s0a = __builtin_amdgcn_wmma_f32_16x16x32_f16(false, qa0, false, kfa,
                                                         (short)0, zc, false, false);
        v8f s0b = __builtin_amdgcn_wmma_f32_16x16x32_f16(false, qa0, false, kfb,
                                                         (short)0, zc, false, false);
        v8f s1a = __builtin_amdgcn_wmma_f32_16x16x32_f16(false, qa1, false, kfa,
                                                         (short)0, zc, false, false);
        v8f s1b = __builtin_amdgcn_wmma_f32_16x16x32_f16(false, qa1, false, kfb,
                                                         (short)0, zc, false, false);
        softmax_update2(s0a, s0b, mrow[0], lrow[0], o00, o01, plds[w][0], mh, nl);
        softmax_update2(s1a, s1b, mrow[1], lrow[1], o10, o11, plds[w][1], mh, nl);

        v16h pa0 = load_p_frag(plds[w][0], lane);
        v16h pa1 = load_p_frag(plds[w][1], lane);
        v16h v0f = load_b_nmajor(vbase + it * 32, NKV, 0, lane);
        v16h v1f = load_b_nmajor(vbase + it * 32, NKV, 16, lane);
        o00 = __builtin_amdgcn_wmma_f32_16x16x32_f16(false, pa0, false, v0f,
                                                     (short)0, o00, false, false);
        o01 = __builtin_amdgcn_wmma_f32_16x16x32_f16(false, pa0, false, v1f,
                                                     (short)0, o01, false, false);
        o10 = __builtin_amdgcn_wmma_f32_16x16x32_f16(false, pa1, false, v0f,
                                                     (short)0, o10, false, false);
        o11 = __builtin_amdgcn_wmma_f32_16x16x32_f16(false, pa1, false, v1f,
                                                     (short)0, o11, false, false);
    }
    {  // tail: keys 768..783 (jt=48); high half of the chunk is padding
        v16h kfa = load_b_nmajor(kbase, DHEAD, 768, lane);
        v8f s0a = __builtin_amdgcn_wmma_f32_16x16x32_f16(false, qa0, false, kfa,
                                                         (short)0, zc, false, false);
        v8f s1a = __builtin_amdgcn_wmma_f32_16x16x32_f16(false, qa1, false, kfa,
                                                         (short)0, zc, false, false);
        softmax_update_tail(s0a, mrow[0], lrow[0], o00, o01, plds[w][0], mh, nl);
        softmax_update_tail(s1a, mrow[1], lrow[1], o10, o11, plds[w][1], mh, nl);

        v16h pa0 = load_p_frag(plds[w][0], lane);
        v16h pa1 = load_p_frag(plds[w][1], lane);
        v16h v0f = load_b_nmajor(vbase + 768, NKV, 0, lane);   // rows 768..799,
        v16h v1f = load_b_nmajor(vbase + 768, NKV, 16, lane);  // pad covers 784..799
        o00 = __builtin_amdgcn_wmma_f32_16x16x32_f16(false, pa0, false, v0f,
                                                     (short)0, o00, false, false);
        o01 = __builtin_amdgcn_wmma_f32_16x16x32_f16(false, pa0, false, v1f,
                                                     (short)0, o01, false, false);
        o10 = __builtin_amdgcn_wmma_f32_16x16x32_f16(false, pa1, false, v0f,
                                                     (short)0, o10, false, false);
        o11 = __builtin_amdgcn_wmma_f32_16x16x32_f16(false, pa1, false, v1f,
                                                     (short)0, o11, false, false);
    }
#pragma unroll
    for (int r = 0; r < 8; ++r) {
        float inv0 = 1.f / lrow[0][r];
        float inv1 = 1.f / lrow[1][r];
        int ia = i0 + r + mh;
        int ib = i0 + 16 + r + mh;
        half_t* opa = obuf + ((size_t)b * NQ + ia) * CC + h * DHEAD;
        half_t* opb = obuf + ((size_t)b * NQ + ib) * CC + h * DHEAD;
        opa[nl]      = (half_t)(o00[r] * inv0);
        opa[16 + nl] = (half_t)(o01[r] * inv0);
        opb[nl]      = (half_t)(o10[r] * inv1);
        opb[16 + nl] = (half_t)(o11[r] * inv1);
    }
}

// ---------------- kernel 5: output projection GEMM (f32 out) --------------
__global__ __launch_bounds__(128) void proj_gemm_kernel(
        const half_t* __restrict__ obuf, const half_t* __restrict__ projw16,
        const float* __restrict__ pbias, float* __restrict__ out) {
    int lane = threadIdx.x & 31;
    int wid  = blockIdx.x * (blockDim.x >> 5) + (threadIdx.x >> 5);
    int b    = wid / (196 * 4);
    int rem  = wid - b * 196 * 4;
    int nt   = rem >> 2;  // n tile (0..195)
    int og   = rem & 3;   // group of 4 o-tiles
    int ot0  = og * 4;
    const half_t* abase = obuf + (size_t)b * NQ * CC;
    v8f acc[4];
#pragma unroll
    for (int ob = 0; ob < 4; ++ob) acc[ob] = (v8f){};
#pragma unroll
    for (int kc = 0; kc < 8; ++kc) {
        v16h a = load_a_rowmajor(abase + kc * 32, nt * 16, CC, lane);
#pragma unroll
        for (int ob = 0; ob < 4; ++ob) {
            v16h bf = load_b_nmajor(projw16 + kc * 32, CC, (ot0 + ob) * 16, lane);
            acc[ob] = __builtin_amdgcn_wmma_f32_16x16x32_f16(false, a, false, bf,
                                                             (short)0, acc[ob], false, false);
        }
    }
    int mh = (lane & 16) ? 8 : 0;
    int nl = lane & 15;
#pragma unroll
    for (int ob = 0; ob < 4; ++ob) {
        int oo = (ot0 + ob) * 16 + nl;
        float pb = pbias[oo];
#pragma unroll
        for (int r = 0; r < 8; ++r) {
            int n = nt * 16 + r + mh;
            out[((size_t)b * NQ + n) * CC + oo] = acc[ob][r] + pb;
        }
    }
}

// ---------------- host launcher ------------------------------------------
extern "C" void kernel_launch(void* const* d_in, const int* in_sizes, int n_in,
                              void* d_out, int out_size, void* d_ws, size_t ws_size,
                              hipStream_t stream) {
    (void)in_sizes; (void)n_in; (void)out_size; (void)ws_size;
    const float* x      = (const float*)d_in[0];
    const float* q_dw_w = (const float*)d_in[3];
    const float* q_dw_b = (const float*)d_in[4];
    const float* q_bn_g = (const float*)d_in[5];
    const float* q_bn_b = (const float*)d_in[6];
    const float* q_bn_m = (const float*)d_in[7];
    const float* q_bn_v = (const float*)d_in[8];
    const float* q_pw_w = (const float*)d_in[9];
    const float* q_pw_b = (const float*)d_in[10];
    const float* k_dw_w = (const float*)d_in[11];
    const float* k_dw_b = (const float*)d_in[12];
    const float* k_bn_g = (const float*)d_in[13];
    const float* k_bn_b = (const float*)d_in[14];
    const float* k_bn_m = (const float*)d_in[15];
    const float* k_bn_v = (const float*)d_in[16];
    const float* k_pw_w = (const float*)d_in[17];
    const float* k_pw_b = (const float*)d_in[18];
    const float* v_dw_w = (const float*)d_in[19];
    const float* v_dw_b = (const float*)d_in[20];
    const float* v_bn_g = (const float*)d_in[21];
    const float* v_bn_b = (const float*)d_in[22];
    const float* v_bn_m = (const float*)d_in[23];
    const float* v_bn_v = (const float*)d_in[24];
    const float* v_pw_w = (const float*)d_in[25];
    const float* v_pw_b = (const float*)d_in[26];
    const float* proj_w = (const float*)d_in[27];
    const float* proj_b = (const float*)d_in[28];
    float* out = (float*)d_out;

    char* ws = (char*)d_ws;
    size_t cur = 0;
    auto take = [&](size_t bytes) -> void* {
        void* p = ws + cur;
        cur = (cur + bytes + 255) & ~(size_t)255;
        return p;
    };
    half_t* Wq16  = (half_t*)take((size_t)CC * CC * 2);
    half_t* Wk16  = (half_t*)take((size_t)CC * CC * 2);
    half_t* Wv16  = (half_t*)take((size_t)CC * CC * 2);
    half_t* PW16  = (half_t*)take((size_t)CC * CC * 2);
    float*  bnp   = (float*)take(6 * CC * 4);
    half_t* tq    = (half_t*)take((size_t)NB * NQ  * CC * 2);   // [b][n][c]
    half_t* tk    = (half_t*)take((size_t)NB * NKV * CC * 2);   // [b][n][c]
    half_t* tv    = (half_t*)take((size_t)NB * NKV * CC * 2);   // [b][n][c]
    half_t* qhb   = (half_t*)take((size_t)NB * CC * NQ  * 2);   // [b][h][i][d]
    half_t* khb   = (half_t*)take((size_t)NB * CC * NKV * 2);   // [b][h][j][d]
    half_t* vhb   = (half_t*)take((size_t)NB * CC * NKV * 2 + 4096);  // [b][h][d][j] + tail pad
    half_t* obuf  = (half_t*)take((size_t)NB * NQ * CC * 2);    // [b][n][c]

    // 1) weight prep
    prep_kernel<<<CC, 256, 0, stream>>>(q_pw_w, k_pw_w, v_pw_w, proj_w,
        q_bn_g, q_bn_b, q_bn_m, q_bn_v, k_bn_g, k_bn_b, k_bn_m, k_bn_v,
        v_bn_g, v_bn_b, v_bn_m, v_bn_v, Wq16, Wk16, Wv16, PW16, bnp);

    // 2) depthwise + BN  (outputs [b][n][c], coalesced)
    dwconv_bn_kernel<<<NB * (NQ / 16), 256, 0, stream>>>(
        x, q_dw_w, q_dw_b, bnp + 0 * CC, bnp + 1 * CC, tq, 1, IW, NQ);
    dwconv_bn_kernel<<<NB * (NKV / 16), 256, 0, stream>>>(
        x, k_dw_w, k_dw_b, bnp + 2 * CC, bnp + 3 * CC, tk, 2, OHW, NKV);
    dwconv_bn_kernel<<<NB * (NKV / 16), 256, 0, stream>>>(
        x, v_dw_w, v_dw_b, bnp + 4 * CC, bnp + 5 * CC, tv, 2, OHW, NKV);

    // 3) pointwise GEMMs (WMMA). q pre-scaled by d^-0.5.
    pw_gemm_kernel<4><<<(NB * 16 * (196 / 4)) / 4, 128, 0, stream>>>(
        Wq16, q_pw_b, tq, qhb, NQ, 1, ATT_SCALE);
    pw_gemm_kernel<7><<<(NB * 16 * (49 / 7)) / 4, 128, 0, stream>>>(
        Wk16, k_pw_b, tk, khb, NKV, 1, 1.0f);
    pw_gemm_kernel<7><<<(NB * 16 * (49 / 7)) / 4, 128, 0, stream>>>(
        Wv16, v_pw_b, tv, vhb, NKV, 0, 1.0f);

    // 4) flash attention (WMMA, 32 queries/wave, 32-key chunks)
    attention_kernel<<<NB * NHEADS * 49, 64, 0, stream>>>(qhb, khb, vhb, obuf);

    // 5) output projection (WMMA, f32 out, 4 o-tiles/wave)
    proj_gemm_kernel<<<(NB * 196 * 4) / 4, 128, 0, stream>>>(obuf, PW16, proj_b, out);
}
